// TorchSpikingConv2dLayer_51367808860569
// MI455X (gfx1250) — matrix operand order repaired and checked by reading, not compiled, not run on hardware
//
#include <hip/hip_runtime.h>

// Spiking Conv2d (3x3, pad 1, Cin=16 -> Cout=64, 64x64) + LIF scan over T=128, fused.
// Implicit-GEMM on v_wmma_f32_16x16x32_f16 (M=16 Cout, N=16 pixels, K=144 padded to 160),
// double-buffered async global->LDS im2col slabs (ASYNCcnt), membrane state in accumulator
// VGPRs across the whole t-loop. Only the 0/1 spikes (134 MB) touch HBM on the output side.

typedef __attribute__((ext_vector_type(16))) _Float16 v16h;
typedef __attribute__((ext_vector_type(8)))  float    v8f;

typedef __attribute__((address_space(1))) int gint_t;   // global int
typedef __attribute__((address_space(3))) int sint_t;   // LDS int

#define T_STEPS   128
#define CIN       16
#define HDIM      64
#define WDIM      64
#define COUT      64
#define KVALID    144                 // CIN*3*3
#define CPITCH    20                  // padded LDS column pitch (18 used)
#define SLABN     (CIN * 3 * CPITCH)  // floats per slab buffer
#define FILLN     (CIN * 3 * 18)      // 864 im2col elements
#define NQ        7                   // ceil(864 / 128) fill items per thread

#if __has_builtin(__builtin_amdgcn_global_load_async_to_lds_b32) && \
    __has_builtin(__builtin_amdgcn_s_wait_asynccnt)
#define USE_ASYNC 1
#else
#define USE_ASYNC 0
#endif

union V16H { v16h v; _Float16 h[16]; };
union V8F  { v8f  v; float    f[8];  };

// B-matrix 32x16 f16 fragment (ISA 7.12.2): halfs j<8 -> K=j+KOFF, j>=8 -> K=16+(j-8)+KOFF (KOFF=8*half),
// N = lane%16. KC/KOFF compile-time so ci/kh/kw fold to immediate ds offsets.
template<int KC, int KOFF>
__device__ __forceinline__ v16h build_b(const float* sl, int ncol) {
  V16H u;
#pragma unroll
  for (int j = 0; j < 16; ++j) {
    const int k = KC * 32 + KOFF + ((j >> 3) << 4) + (j & 7);
    if (k < KVALID) {
      const int ci = k / 9;
      const int rr = (k % 9) / 3;
      const int cc = k % 3;
      u.h[j] = (_Float16)sl[(ci * 3 + rr) * CPITCH + ncol + cc];
    } else {
      u.h[j] = (_Float16)0.f;
    }
  }
  return u.v;
}

__global__ __launch_bounds__(128)
void spiking_conv_wmma_kernel(const float* __restrict__ x,
                              const float* __restrict__ Wt,
                              const float* __restrict__ bias,
                              float* __restrict__ out) {
  __shared__ float slabA[SLABN];      // double-buffered f32 im2col slabs (~7.5 KB total)
  __shared__ float slabB[SLABN];

  const int tid  = threadIdx.x;
  const int lane = tid & 31;
  const int wave = tid >> 5;          // 0..3 -> C_out tile
  const int half = lane >> 4;         // 0/1 (lane group)
  const int ncol = lane & 15;         // N (pixel) / M (co) sub-index

  const int bx = blockIdx.x;          // 256 blocks = 64 rows x 4 strips
  const int h  = bx >> 2;             // output row
  const int w0 = (bx & 3) * 16;       // output col base

  const int co_base = wave * 16;
  const int co_half = co_base + half * 8;   // C/D layout: VGPR r -> M = r + 8*half

  // ---- Per-thread fill descriptors (t-invariant) + one-time border zeroing ----
  int loff[NQ];   // LDS float offset, -1 = nothing to fetch
  int goff[NQ];   // global float offset within x[t] slice
#pragma unroll
  for (int q = 0; q < NQ; ++q) {
    const int i = tid + q * 128;
    loff[q] = -1; goff[q] = 0;
    if (i < FILLN) {
      const int ci  = i / 54;
      const int rem = i % 54;
      const int rr  = rem / 18;
      const int cc  = rem % 18;
      const int row = h  + rr - 1;
      const int col = w0 + cc - 1;
      const int lo  = (ci * 3 + rr) * CPITCH + cc;
      if ((unsigned)row < (unsigned)HDIM && (unsigned)col < (unsigned)WDIM) {
        loff[q] = lo;
        goff[q] = (ci * HDIM + row) * WDIM + col;
      } else {
        slabA[lo] = 0.f;              // static zero padding: never overwritten
        slabB[lo] = 0.f;
      }
    }
  }

  // Issue the im2col fill of one slab buffer for one timestep (async on CDNA5).
  auto issue_fill = [&](float* sb, const float* xt) {
#pragma unroll
    for (int q = 0; q < NQ; ++q) {
      if (loff[q] >= 0) {
#if USE_ASYNC
        __builtin_amdgcn_global_load_async_to_lds_b32(
            (gint_t*)(xt + goff[q]),
            (sint_t*)(sb + loff[q]),
            0, 0);
#else
        sb[loff[q]] = xt[goff[q]];
#endif
      }
    }
  };

  // ---- A fragments (weights f32->f16), 16x32 per chunk, invariant over t ----
  // A layout (ISA 7.12.2, 16-bit A 16x32): halfs j<8 -> K = j + 8*half,
  //                                        halfs j>=8 -> K = 16 + (j-8) + 8*half.
  V16H a[5];
#pragma unroll
  for (int kc = 0; kc < 5; ++kc) {
#pragma unroll
    for (int j = 0; j < 16; ++j) {
      const int K = kc * 32 + ((j >> 3) << 4) + (j & 7) + half * 8;
      a[kc].h[j] = (K < KVALID)
                 ? (_Float16)Wt[(co_base + ncol) * KVALID + K]
                 : (_Float16)0.f;
    }
  }

  // ---- bias fragment + membrane state ----
  V8F bfrag, state;
#pragma unroll
  for (int r = 0; r < 8; ++r) { bfrag.f[r] = bias[co_half + r]; state.f[r] = 0.f; }

  // ---- prologue: fill slab A for t=0 ----
  issue_fill(slabA, x);
#if USE_ASYNC
  __builtin_amdgcn_s_wait_asynccnt(0);
#endif
  __syncthreads();

  for (int t = 0; t < T_STEPS; ++t) {
    const int cur = t & 1;
    // kick off next slab while we compute this one (disjoint buffer)
    if (t + 1 < T_STEPS)
      issue_fill(cur ? slabA : slabB, x + (size_t)(t + 1) * (CIN * HDIM * WDIM));

    const float* sl = cur ? slabB : slabA;

    // ---- B fragments (divergent on lane-half, reconverges before WMMA) ----
    v16h b0, b1, b2, b3, b4;
    if (half == 0) {
      b0 = build_b<0, 0>(sl, ncol);  b1 = build_b<1, 0>(sl, ncol);
      b2 = build_b<2, 0>(sl, ncol);  b3 = build_b<3, 0>(sl, ncol);
      b4 = build_b<4, 0>(sl, ncol);
    } else {
      b0 = build_b<0, 8>(sl, ncol); b1 = build_b<1, 8>(sl, ncol);
      b2 = build_b<2, 8>(sl, ncol); b3 = build_b<3, 8>(sl, ncol);
      b4 = build_b<4, 8>(sl, ncol);
    }

    // ---- conv_t = bias + A x B  (5 chained K=32 WMMAs, f32 accumulate) ----
    V8F conv; conv.v = bfrag.v;
    conv.v = __builtin_amdgcn_wmma_f32_16x16x32_f16(false, a[0].v, false, b0, (short)0, conv.v, false, false);
    conv.v = __builtin_amdgcn_wmma_f32_16x16x32_f16(false, a[1].v, false, b1, (short)0, conv.v, false, false);
    conv.v = __builtin_amdgcn_wmma_f32_16x16x32_f16(false, a[2].v, false, b2, (short)0, conv.v, false, false);
    conv.v = __builtin_amdgcn_wmma_f32_16x16x32_f16(false, a[3].v, false, b3, (short)0, conv.v, false, false);
    conv.v = __builtin_amdgcn_wmma_f32_16x16x32_f16(false, a[4].v, false, b4, (short)0, conv.v, false, false);

    // ---- LIF update in registers + spike store ----
    float* op = out + (((size_t)t * COUT + co_half) * (HDIM * WDIM))
                    + h * WDIM + w0 + ncol;
#pragma unroll
    for (int r = 0; r < 8; ++r) {
      float s = state.f[r] + conv.f[r];
      const bool fire = (s >= 8.0f);
      op[r * (HDIM * WDIM)] = fire ? 1.0f : 0.0f;
      s = fire ? 0.0f : s;               // reset-to-value (0)
      state.f[r] = fmaxf(s, -1.0f);      // nn.Threshold(-1, -1)
    }

    // next-slab fill must be complete & visible to all waves before t+1 reads it
#if USE_ASYNC
    __builtin_amdgcn_s_wait_asynccnt(0);
#endif
    __syncthreads();
  }
}

extern "C" void kernel_launch(void* const* d_in, const int* in_sizes, int n_in,
                              void* d_out, int out_size, void* d_ws, size_t ws_size,
                              hipStream_t stream) {
  (void)in_sizes; (void)n_in; (void)out_size; (void)d_ws; (void)ws_size;
  const float* x  = (const float*)d_in[0];   // [128,16,64,64]
  const float* W  = (const float*)d_in[1];   // [64,16,3,3]
  const float* b  = (const float*)d_in[2];   // [64]
  float* out = (float*)d_out;                // [128,64,64,64]
  spiking_conv_wmma_kernel<<<256, 128, 0, stream>>>(x, W, b, out);
}
